// GCN_MME_77506979823983
// MI455X (gfx1250) — matrix operand order repaired and checked by reading, not compiled, not run on hardware
//
#include <hip/hip_runtime.h>

// ---------------------------------------------------------------------------
// GCN-MME for MI455X (gfx1250): WMMA f16 encoder GEMMs + atomic graph conv
// ---------------------------------------------------------------------------

typedef _Float16 half4_t __attribute__((ext_vector_type(4)));
typedef _Float16 half8_t __attribute__((ext_vector_type(8)));
typedef _Float16 v16h    __attribute__((ext_vector_type(16)));
typedef float    v8f     __attribute__((ext_vector_type(8)));
typedef float    floatx4 __attribute__((ext_vector_type(4)));

#define BM 128
#define BK 32

// problem sizes (fixed by the reference)
#define NN    50000
#define EE    1600000
#define MMOD  2
#define DIN   2000
#define H1    500
#define H1P   512    // padded h1 leading dim / layer-1 out cols (multiple of 128)
#define K1P   2016   // padded K for layer-1 weights (63*32)
#define LATD  128
#define DECD  64
#define HIDD  64
#define NCLS  16

// ---------------------------------------------------------------------------
// helper kernels
// ---------------------------------------------------------------------------

// f32 [nm, K, C] -> f16 TRANSPOSED [nm, Cpad, Kpad], zero-padded
__global__ void convert_pad_tr_kernel(const float* __restrict__ in, _Float16* __restrict__ out,
                                      int K, int C, int Kpad, int Cpad, int nm) {
    long long t = (long long)blockIdx.x * blockDim.x + threadIdx.x;
    long long total = (long long)nm * Cpad * Kpad;
    if (t < total) {
        int per = Cpad * Kpad;
        int m = (int)(t / per);
        int rem = (int)(t % per);
        int c = rem / Kpad;
        int k = rem % Kpad;
        _Float16 v = (_Float16)0.f;
        if (k < K && c < C) v = (_Float16)in[((long long)m * K + k) * C + c];
        out[t] = v;
    }
}

// scale = g * rsqrt(v + eps); shift = (b - mu) * scale + be   (inference BN fold)
__global__ void bn_fold_kernel(const float* __restrict__ b, const float* __restrict__ g,
                               const float* __restrict__ be, const float* __restrict__ mu,
                               const float* __restrict__ v, float* __restrict__ scale,
                               float* __restrict__ shift, int n) {
    int t = blockIdx.x * blockDim.x + threadIdx.x;
    if (t < n) {
        float s = g[t] * __frsqrt_rn(v[t] + 1e-5f);
        scale[t] = s;
        shift[t] = (b[t] - mu[t]) * s + be[t];
    }
}

// layer-3 epilogue fold: mean over modalities -> scale 0.5, shift 0.5*(bd0+bd1)
__global__ void fold3_kernel(const float* __restrict__ bd, float* __restrict__ scale,
                             float* __restrict__ shift, int n) {
    int t = blockIdx.x * blockDim.x + threadIdx.x;
    if (t < n) {
        scale[t] = 0.5f;
        shift[t] = 0.5f * (bd[t] + bd[n + t]);
    }
}

// ---------------------------------------------------------------------------
// WMMA tiled GEMM, double-buffered LDS, branch-free staging:
//   Out[m] = (A[m] @ B[m]) * scale[m] + shift[m]
// A: row-major [Nrows x lda], AT = float (converted in regs) or f16 (padded,
//    Kdim % 32 == 0 guaranteed by caller).
// B: PRE-TRANSPOSED zero-padded f16 [Cpad x Kpad] -> both the global load and
//    the LDS store of the B tile are contiguous 16B accesses (no scalar
//    transpose stores), and each 16x16 B fragment is one 32B LDS read.
// NMOD==2: both modalities accumulate into the same accumulators (layer 3).
// Block: 256 threads = 8 waves (4x2), tile 128 x BNt, wave tile 32 x (BNt/2).
// ---------------------------------------------------------------------------
template <typename AT, int NMOD, bool HALF_OUT, int BNt>
__global__ void __launch_bounds__(256)
gemm_bn_kernel(const AT* __restrict__ Abase, long long strideAm, int lda,
               const _Float16* __restrict__ Btr, long long strideBm, int ldb /* = Kpad */,
               const float* __restrict__ scale, const float* __restrict__ shift,
               int scaleStride,
               void* __restrict__ Out, long long strideOm, int ldo, int storeCols,
               int Nrows, int Kdim, int Ncols) {
    constexpr int NR = BNt / 32;                 // WMMA col subtiles per wave (2 or 4)
    constexpr int BITER = (BNt * BK / 8) / 256;  // B-stage chunks per thread (1 or 2)

    __shared__ __align__(32) _Float16 lA[2][BM * BK];    // 2 x 8 KB
    __shared__ __align__(32) _Float16 lB[2][BNt * BK];   // 2 x (BNt/16) KB, [col][k]

    const int tid  = threadIdx.x;
    const int lane = tid & 31;
    const int wave = tid >> 5;          // 0..7 (wave32)
    const int waveRow = wave >> 1;      // 0..3  (32-row strips)
    const int waveCol = wave & 1;       // 0..1  (NR*16-col strips)
    const int blockRow = blockIdx.x;
    const int blockCol = blockIdx.y;
    const int zm = (NMOD == 2) ? 0 : blockIdx.z;
    const int nT = (Kdim + BK - 1) / BK;

    v8f acc[2][NR] = {};

    // prefetch registers
    floatx4 aF[4];
    half8_t aH[2];
    half8_t bR[BITER];
    const AT* Am = nullptr;
    const _Float16* Bm = nullptr;

    auto loadA = [&](int k0) {
        if constexpr (sizeof(AT) == 4) {
#pragma unroll
            for (int i = 0; i < 4; ++i) {
                int idx = tid + i * 256;                // 0..1023
                int r   = idx >> 3;                     // 0..127
                int c4  = (idx & 7) << 2;               // 0..28
                long long grow = (long long)blockRow * BM + r;
                if (grow >= Nrows) grow = Nrows - 1;    // clamp: always legal
                int gk = k0 + c4;
                bool ok = (gk + 4 <= Kdim);
                int gks = ok ? gk : 0;
                floatx4 f = *(const floatx4*)((const float*)Am + grow * (long long)lda + gks);
                if (!ok) f = (floatx4){0.f, 0.f, 0.f, 0.f};
                aF[i] = f;
            }
        } else {
#pragma unroll
            for (int i = 0; i < 2; ++i) {
                int idx = tid + i * 256;                // 0..511
                int r   = idx >> 2;                     // 0..127
                int c8  = (idx & 3) << 3;               // 0,8,16,24
                long long grow = (long long)blockRow * BM + r;
                if (grow >= Nrows) grow = Nrows - 1;
                aH[i] = *(const half8_t*)((const _Float16*)Am + grow * (long long)lda + k0 + c8);
            }
        }
    };
    auto storeA = [&](int buf) {
        if constexpr (sizeof(AT) == 4) {
#pragma unroll
            for (int i = 0; i < 4; ++i) {
                int idx = tid + i * 256;
                int r   = idx >> 3;
                int c4  = (idx & 7) << 2;
                half4_t h4 = {(_Float16)aF[i][0], (_Float16)aF[i][1],
                              (_Float16)aF[i][2], (_Float16)aF[i][3]};
                *(half4_t*)(&lA[buf][r * BK + c4]) = h4;
            }
        } else {
#pragma unroll
            for (int i = 0; i < 2; ++i) {
                int idx = tid + i * 256;
                int r   = idx >> 2;
                int c8  = (idx & 3) << 3;
                *(half8_t*)(&lA[buf][r * BK + c8]) = aH[i];
            }
        }
    };
    auto loadB = [&](int k0) {
#pragma unroll
        for (int i = 0; i < BITER; ++i) {
            int chunk = tid + i * 256;               // 0..BNt*4-1
            int c  = chunk >> 2;                     // 0..BNt-1
            int k8 = (chunk & 3) << 3;               // 0,8,16,24
            bR[i] = *(const half8_t*)(Bm + (long long)c * ldb + k0 + k8);
        }
    };
    auto storeB = [&](int buf) {
#pragma unroll
        for (int i = 0; i < BITER; ++i) {
            int chunk = tid + i * 256;
            int c  = chunk >> 2;
            int k8 = (chunk & 3) << 3;
            *(half8_t*)(&lB[buf][c * BK + k8]) = bR[i];
        }
    };
    auto compute = [&](int buf) {
        const _Float16* As = lA[buf];
        const _Float16* Bs = lB[buf];
        // A fragment: lanes 0-15 row=lane, K 0-15;
        //             lanes 16-31 row=lane-16, K 16-31
        v16h af[2], bf[NR];
#pragma unroll
        for (int r = 0; r < 2; ++r) {
            int row  = waveRow * 32 + r * 16 + (lane & 15);
            int koff = (lane >> 4) * 16;
            half8_t lo = *(const half8_t*)(As + row * BK + koff);
            half8_t hi = *(const half8_t*)(As + row * BK + koff + 8);
            af[r] = __builtin_shufflevector(lo, hi, 0, 1, 2, 3, 4, 5, 6, 7,
                                            8, 9, 10, 11, 12, 13, 14, 15);
        }
        // B fragment: col = lane&15, lanes 0-15 K 0-15, lanes 16-31 K 16-31
#pragma unroll
        for (int c = 0; c < NR; ++c) {
            int col = waveCol * (NR * 16) + c * 16 + (lane & 15);
            int kb  = (lane >> 4) * 16;
            bf[c] = *(const v16h*)(Bs + col * BK + kb);
        }
#pragma unroll
        for (int r = 0; r < 2; ++r)
#pragma unroll
            for (int c = 0; c < NR; ++c)
                acc[r][c] = __builtin_amdgcn_wmma_f32_16x16x32_f16(
                    false, af[r], false, bf[c], (short)0, acc[r][c], false, false);
    };

    for (int mm = 0; mm < NMOD; ++mm) {
        const int mSel = (NMOD == 2) ? mm : (int)blockIdx.z;
        Am = Abase + (long long)mSel * strideAm;
        Bm = Btr + (long long)mSel * strideBm + (long long)blockCol * BNt * ldb;

        loadA(0);
        loadB(0);
        storeA(0);
        storeB(0);
        __syncthreads();
        for (int t = 0; t < nT; ++t) {
            int cur = t & 1;
            bool more = (t + 1 < nT);
            if (more) { loadA((t + 1) * BK); loadB((t + 1) * BK); }
            compute(cur);
            __syncthreads();
            if (more) {
                storeA(cur ^ 1);
                storeB(cur ^ 1);
                __syncthreads();
            }
        }
    }

    // ---- fused affine epilogue (folded BN / bias / modality mean) ----
    const float* sc = scale + zm * scaleStride;
    const float* sh = shift + zm * scaleStride;
#pragma unroll
    for (int r = 0; r < 2; ++r) {
#pragma unroll
        for (int c = 0; c < NR; ++c) {
            int gcol = blockCol * BNt + waveCol * (NR * 16) + c * 16 + (lane & 15);
            if (gcol >= storeCols) continue;
            bool okc = (gcol < Ncols);
            int ci = okc ? gcol : 0;
            float s = sc[ci];
            float t = sh[ci];
            long long rbase = (long long)blockRow * BM + waveRow * 32 + r * 16 +
                              ((lane >> 4) << 3);
#pragma unroll
            for (int i = 0; i < 8; ++i) {
                long long grow = rbase + i;
                if (grow < Nrows) {
                    float val = okc ? (acc[r][c][i] * s + t) : 0.0f;  // zero-fill pad cols
                    if (HALF_OUT) {
                        ((_Float16*)Out)[(long long)zm * strideOm + grow * ldo + gcol] =
                            (_Float16)val;
                    } else {
                        ((float*)Out)[(long long)zm * strideOm + grow * ldo + gcol] = val;
                    }
                }
            }
        }
    }
}

// ---------------------------------------------------------------------------
// graph kernels
// ---------------------------------------------------------------------------
__global__ void degree_kernel(const int* __restrict__ src, const int* __restrict__ dst,
                              float* __restrict__ dout, float* __restrict__ din, int E) {
    int t = blockIdx.x * blockDim.x + threadIdx.x;
    if (t < E) {
        atomicAdd(&dout[src[t]], 1.0f);
        atomicAdd(&din[dst[t]], 1.0f);
    }
}

// msg = (feat * deg_out^-1/2)[src]; agg[dst] += msg   (F = 64 features)
__global__ void scatter_kernel(const float* __restrict__ feat, const int* __restrict__ src,
                               const int* __restrict__ dst, const float* __restrict__ deg_out,
                               float* __restrict__ agg, int E) {
    int t = blockIdx.x * blockDim.x + threadIdx.x;
    if (t < E * 64) {
        int e = t >> 6;
        int f = t & 63;
        int s = src[e];
        float d = deg_out[s];
        float cs = __frsqrt_rn(d < 1.0f ? 1.0f : d);
        float v = feat[(long long)s * 64 + f] * cs;
        atomicAdd(&agg[(long long)dst[e] * 64 + f], v);
    }
}

// out[n] = act( (agg[n] * deg_in[n]^-1/2) @ W + b )
template <int IN, int OUT, bool RELU>
__global__ void __launch_bounds__(256)
dense_kernel(const float* __restrict__ agg, const float* __restrict__ deg_in,
             const float* __restrict__ W, const float* __restrict__ bias,
             float* __restrict__ out, int Nn) {
    __shared__ float lW[IN * OUT];
    __shared__ float lb[OUT];
    for (int i = threadIdx.x; i < IN * OUT; i += 256) lW[i] = W[i];
    if (threadIdx.x < OUT) lb[threadIdx.x] = bias[threadIdx.x];
    __syncthreads();

    const int NPB = 256 / OUT;
    long long node = (long long)blockIdx.x * NPB + threadIdx.x / OUT;
    int o = threadIdx.x % OUT;
    if (node < Nn) {
        float d = deg_in[node];
        float cd = __frsqrt_rn(d < 1.0f ? 1.0f : d);
        float sum = 0.0f;
        const float* fr = agg + node * IN;
#pragma unroll 8
        for (int k = 0; k < IN; ++k) sum += fr[k] * lW[k * OUT + o];
        float val = sum * cd + lb[o];
        if (RELU) val = val > 0.0f ? val : 0.0f;
        out[node * OUT + o] = val;
    }
}

// ---------------------------------------------------------------------------
// host orchestration
// ---------------------------------------------------------------------------
extern "C" void kernel_launch(void* const* d_in, const int* in_sizes, int n_in,
                              void* d_out, int out_size, void* d_ws, size_t ws_size,
                              hipStream_t stream) {
    (void)in_sizes; (void)n_in; (void)out_size; (void)ws_size;

    const float* x   = (const float*)d_in[0];   // [M,N,DIN]
    const int*   src = (const int*)d_in[1];     // [E]
    const int*   dst = (const int*)d_in[2];     // [E]
    const float* W1  = (const float*)d_in[3];   // [M,DIN,H1]
    const float* b1  = (const float*)d_in[4];
    const float* g1  = (const float*)d_in[5];
    const float* be1 = (const float*)d_in[6];
    const float* m1  = (const float*)d_in[7];
    const float* v1  = (const float*)d_in[8];
    const float* W2  = (const float*)d_in[9];   // [M,H1,LAT]
    const float* b2  = (const float*)d_in[10];
    const float* g2  = (const float*)d_in[11];
    const float* be2 = (const float*)d_in[12];
    const float* m2  = (const float*)d_in[13];
    const float* v2  = (const float*)d_in[14];
    const float* Wd  = (const float*)d_in[15];  // [M,LAT,DEC]
    const float* bd  = (const float*)d_in[16];  // [M,DEC]
    const float* Wg1 = (const float*)d_in[17];  // [DEC,HID]
    const float* bg1 = (const float*)d_in[18];
    const float* Wg2 = (const float*)d_in[19];  // [HID,NC]
    const float* bg2 = (const float*)d_in[20];

    // ---- workspace layout (256B aligned) ----
    char* base = (char*)d_ws;
    size_t off = 0;
    auto take = [&](size_t bytes) -> char* {
        char* p = base + off;
        off = (off + bytes + 255) & ~(size_t)255;
        return p;
    };
    _Float16* W1t    = (_Float16*)take((size_t)MMOD * H1P * K1P * 2);   // [M,512,2016] f16^T
    _Float16* W2t    = (_Float16*)take((size_t)MMOD * LATD * H1P * 2);  // [M,128,512]  f16^T
    _Float16* Wdt    = (_Float16*)take((size_t)MMOD * DECD * LATD * 2); // [M,64,128]   f16^T
    float*    scale1 = (float*)take((size_t)MMOD * H1 * 4);
    float*    shift1 = (float*)take((size_t)MMOD * H1 * 4);
    float*    scale2 = (float*)take((size_t)MMOD * LATD * 4);
    float*    shift2 = (float*)take((size_t)MMOD * LATD * 4);
    float*    scale3 = (float*)take((size_t)DECD * 4);
    float*    shift3 = (float*)take((size_t)DECD * 4);
    _Float16* h1     = (_Float16*)take((size_t)MMOD * NN * H1P * 2);    // padded lda
    _Float16* h2     = (_Float16*)take((size_t)MMOD * NN * LATD * 2);
    float*    nf     = (float*)take((size_t)NN * DECD * 4);
    float*    degO   = (float*)take((size_t)NN * 4);
    float*    degI   = (float*)take((size_t)NN * 4);
    float*    agg1   = (float*)take((size_t)NN * 64 * 4);
    float*    hfeat  = (float*)take((size_t)NN * 64 * 4);
    float*    agg2   = (float*)take((size_t)NN * 64 * 4);

    // ---- weight conversion (zero-padded, transposed) + BN folding ----
    {
        long long n = (long long)MMOD * H1P * K1P;
        convert_pad_tr_kernel<<<(unsigned)((n + 255) / 256), 256, 0, stream>>>(
            W1, W1t, DIN, H1, K1P, H1P, MMOD);
        n = (long long)MMOD * LATD * H1P;
        convert_pad_tr_kernel<<<(unsigned)((n + 255) / 256), 256, 0, stream>>>(
            W2, W2t, H1, LATD, H1P, LATD, MMOD);
        n = (long long)MMOD * DECD * LATD;
        convert_pad_tr_kernel<<<(unsigned)((n + 255) / 256), 256, 0, stream>>>(
            Wd, Wdt, LATD, DECD, LATD, DECD, MMOD);
    }
    bn_fold_kernel<<<(MMOD * H1 + 255) / 256, 256, 0, stream>>>(b1, g1, be1, m1, v1,
                                                                scale1, shift1, MMOD * H1);
    bn_fold_kernel<<<(MMOD * LATD + 255) / 256, 256, 0, stream>>>(b2, g2, be2, m2, v2,
                                                                  scale2, shift2, MMOD * LATD);
    fold3_kernel<<<1, 64, 0, stream>>>(bd, scale3, shift3, DECD);

    // ---- encoder GEMMs (WMMA) ----
    const int rowBlocks = (NN + BM - 1) / BM;  // 391
    // layer 1: [M] x (N x 2000) @ (2000 x 500) -> h1 f16 [M,N,512] (cols 500..511 zeroed)
    gemm_bn_kernel<float, 1, true, 128><<<dim3(rowBlocks, H1P / 128, MMOD), 256, 0, stream>>>(
        x, (long long)NN * DIN, DIN,
        W1t, (long long)H1P * K1P, K1P,
        scale1, shift1, H1,
        h1, (long long)NN * H1P, H1P, H1P,
        NN, DIN, H1);
    // layer 2: [M] x (N x 512p) @ (512p x 128) -> h2 f16 [M,N,128]
    gemm_bn_kernel<_Float16, 1, true, 128><<<dim3(rowBlocks, 1, MMOD), 256, 0, stream>>>(
        h1, (long long)NN * H1P, H1P,
        W2t, (long long)LATD * H1P, H1P,
        scale2, shift2, LATD,
        h2, (long long)NN * LATD, LATD, LATD,
        NN, H1P, LATD);
    // layer 3 + modality mean: 0.5 * sum_m (N x 128) @ (128 x 64) -> nf f32 [N,64]
    gemm_bn_kernel<_Float16, 2, false, 64><<<dim3(rowBlocks, 1, 1), 256, 0, stream>>>(
        h2, (long long)NN * LATD, LATD,
        Wdt, (long long)DECD * LATD, LATD,
        scale3, shift3, DECD,
        nf, 0, DECD, DECD,
        NN, LATD, DECD);

    // ---- graph conv ----
    hipMemsetAsync(degO, 0, (size_t)NN * 4, stream);
    hipMemsetAsync(degI, 0, (size_t)NN * 4, stream);
    hipMemsetAsync(agg1, 0, (size_t)NN * 64 * 4, stream);
    hipMemsetAsync(agg2, 0, (size_t)NN * 64 * 4, stream);

    degree_kernel<<<(EE + 255) / 256, 256, 0, stream>>>(src, dst, degO, degI, EE);

    scatter_kernel<<<(EE * 64) / 256, 256, 0, stream>>>(nf, src, dst, degO, agg1, EE);
    dense_kernel<DECD, HIDD, true><<<NN / (256 / HIDD), 256, 0, stream>>>(
        agg1, degI, Wg1, bg1, hfeat, NN);

    scatter_kernel<<<(EE * 64) / 256, 256, 0, stream>>>(hfeat, src, dst, degO, agg2, EE);
    dense_kernel<HIDD, NCLS, false><<<NN / (256 / NCLS), 256, 0, stream>>>(
        agg2, degI, Wg2, bg2, (float*)d_out, NN);
}
